// CoarseMVSNet_65438121722297
// MI455X (gfx1250) — matrix operand order repaired and checked by reading, not compiled, not run on hardware
//
#include <hip/hip_runtime.h>
#include <hip/hip_fp16.h>
#include <cstdint>
#include <cstddef>

// ---------------------------------------------------------------------------
// CDNA5 / gfx1250 implementation of the two-stage MVSNet reference.
//  - f16 variance volume (halves dominant traffic; 63MB fits in 192MB L2)
//  - conv3d (C->1, 3x3x3) as implicit GEMM on v_wmma_f32_16x16x32_f16:
//    M = 16 consecutive W voxels, K = channels, 27 accumulating WMMAs.
//  - weights pre-packed to f16 in a [tap][K(pad32)] layout so the B fragment
//    is two unpredicated 16B vector loads per tap (no scalar-load serialization).
//  - conv + warp kernels templated on C so all guards are compile-time.
// ---------------------------------------------------------------------------

typedef __attribute__((ext_vector_type(16))) _Float16     v16h;
typedef __attribute__((ext_vector_type(8)))  float        v8f;
typedef __attribute__((ext_vector_type(4)))  unsigned int u32x4;

#define ND1 48
#define ND2 8
#define LAMB 1.5f

// ------------------------------ small math --------------------------------

__device__ __forceinline__ void mat4_compose(const float* __restrict__ p, float* o) {
  // p: [0]=E (4x4), [1]=Kin (4x4). out = E with top 3 rows = Kin[:3,:3] @ E[:3,:4]
  const float* E = p;
  const float* K = p + 16;
  for (int r = 0; r < 3; ++r)
    for (int c = 0; c < 4; ++c) {
      float s = 0.f;
      for (int k = 0; k < 3; ++k) s += K[r * 4 + k] * E[k * 4 + c];
      o[r * 4 + c] = s;
    }
  for (int c = 0; c < 4; ++c) o[12 + c] = E[12 + c];  // [0,0,0,1]
}

__device__ __forceinline__ void mat4_affine_inv(const float* m, float* o) {
  // m = [[A, b],[0,0,0,1]] ; inverse = [[A^-1, -A^-1 b],[0,0,0,1]]
  float a00 = m[0], a01 = m[1], a02 = m[2];
  float a10 = m[4], a11 = m[5], a12 = m[6];
  float a20 = m[8], a21 = m[9], a22 = m[10];
  float b0 = m[3], b1 = m[7], b2 = m[11];
  float c00 = a11 * a22 - a12 * a21;
  float c01 = -(a10 * a22 - a12 * a20);
  float c02 = a10 * a21 - a11 * a20;
  float det = a00 * c00 + a01 * c01 + a02 * c02;
  float id = 1.f / det;
  float i00 = c00 * id, i01 = -(a01 * a22 - a02 * a21) * id, i02 = (a01 * a12 - a02 * a11) * id;
  float i10 = c01 * id, i11 = (a00 * a22 - a02 * a20) * id, i12 = -(a00 * a12 - a02 * a10) * id;
  float i20 = c02 * id, i21 = -(a00 * a21 - a01 * a20) * id, i22 = (a00 * a11 - a01 * a10) * id;
  o[0] = i00; o[1] = i01; o[2] = i02; o[3] = -(i00 * b0 + i01 * b1 + i02 * b2);
  o[4] = i10; o[5] = i11; o[6] = i12; o[7] = -(i10 * b0 + i11 * b1 + i12 * b2);
  o[8] = i20; o[9] = i21; o[10] = i22; o[11] = -(i20 * b0 + i21 * b1 + i22 * b2);
  o[12] = 0.f; o[13] = 0.f; o[14] = 0.f; o[15] = 1.f;
}

__device__ __forceinline__ void mat4_mul(const float* a, const float* b, float* o) {
  for (int r = 0; r < 4; ++r)
    for (int c = 0; c < 4; ++c) {
      float s = 0.f;
      for (int k = 0; k < 4; ++k) s += a[r * 4 + k] * b[k * 4 + c];
      o[r * 4 + c] = s;
    }
}

// jax.image.resize 'bilinear' upsample x2 (half-pixel convention, edge clamp)
__device__ __forceinline__ float bilin2d(const float* __restrict__ src, int Hs, int Ws,
                                         int h, int w) {
  float sy = (h + 0.5f) * 0.5f - 0.5f;
  float sx = (w + 0.5f) * 0.5f - 0.5f;
  float y0f = floorf(sy), x0f = floorf(sx);
  float fy = sy - y0f, fx = sx - x0f;
  int y0 = (int)y0f, x0 = (int)x0f;
  int y0c = min(max(y0, 0), Hs - 1), y1c = min(max(y0 + 1, 0), Hs - 1);
  int x0c = min(max(x0, 0), Ws - 1), x1c = min(max(x0 + 1, 0), Ws - 1);
  float v00 = src[y0c * Ws + x0c], v01 = src[y0c * Ws + x1c];
  float v10 = src[y1c * Ws + x0c], v11 = src[y1c * Ws + x1c];
  return v00 * (1.f - fx) * (1.f - fy) + v01 * fx * (1.f - fy) +
         v10 * (1.f - fx) * fy + v11 * fx * fy;
}

// depth sample(d) = s0 + sstep*d (affine in d for both stages)
template <int STAGE>
__device__ __forceinline__ void samp_base_step(int h, int w,
                                               const float* __restrict__ dv,
                                               const float* __restrict__ d1,
                                               const float* __restrict__ v1,
                                               int H1, int W1, float& s0, float& sstep) {
  if (STAGE == 1) {
    float dmin = dv[0], dmax = dv[ND1 - 1];
    sstep = (dmax - dmin) * (1.f / (float)(ND1 - 1));
    s0 = dmin;
  } else {
    float cd = bilin2d(d1, H1, W1, h, w);
    float cv = bilin2d(v1, H1, W1, h, w);
    float low = -fminf(cd, cv);
    sstep = (cv - low) * (1.f / (float)(ND2 - 1));
    s0 = cd + low + 1e-12f;
  }
}

// --------------------------- K0: compose projections -----------------------

__global__ void mvs_compose_kernel(const float* __restrict__ proj1,
                                   const float* __restrict__ proj2,
                                   float* __restrict__ rt) {
  int t = threadIdx.x;
  if (t >= 4) return;
  int stage = t >> 1;  // 0 -> stage1, 1 -> stage2
  int vi = t & 1;      // source view index (view = vi+1)
  const float* projs = stage ? proj2 : proj1;  // (3 views, 2, 4, 4)
  float refC[16], refI[16], srcC[16], P[16];
  mat4_compose(projs, refC);
  mat4_affine_inv(refC, refI);
  mat4_compose(projs + (size_t)(vi + 1) * 32, srcC);
  mat4_mul(srcC, refI, P);
  float* o = rt + (size_t)(stage * 2 + vi) * 12;
  o[0] = P[0]; o[1] = P[1]; o[2] = P[2];
  o[3] = P[4]; o[4] = P[5]; o[5] = P[6];
  o[6] = P[8]; o[7] = P[9]; o[8] = P[10];
  o[9] = P[3]; o[10] = P[7]; o[11] = P[11];
}

// ---------------- K0b: pack conv weights to f16, WMMA layout ----------------
// wgtH[stage*27*32 + tap*32 + K], K zero-padded to 32. Lets the conv kernel
// build its B fragment with two unpredicated 16B vector loads per tap.

__global__ void mvs_pack_weights_kernel(const float* __restrict__ w1,
                                        const float* __restrict__ w2,
                                        _Float16* __restrict__ wgtH) {
  int t = blockIdx.x * blockDim.x + threadIdx.x;
  if (t >= 2 * 27 * 32) return;
  int stage = t / (27 * 32);
  int r = t - stage * 27 * 32;
  int tap = r >> 5;
  int K = r & 31;
  const float* w = stage ? w2 : w1;
  int Cs = stage ? 16 : 32;
  wgtH[t] = (_Float16)((K < Cs) ? w[K * 27 + tap] : 0.f);
}

// ------------------- K1: fused homography warp + variance ------------------
// One thread per voxel (d,h,w); loops C channels; writes f16 variance volume
// in channel-innermost layout var[((d*H+h)*W+w)*C + c]; row stored with C/8
// 16-byte vector stores.

template <int C, int STAGE>
__global__ void mvs_warp_var_kernel(const float* __restrict__ feats,  // (3,C,H,W)
                                    const float* __restrict__ rt,     // 2 views x 12
                                    const float* __restrict__ dv,
                                    const float* __restrict__ d1,
                                    const float* __restrict__ v1,
                                    _Float16* __restrict__ varBuf,
                                    int D, int H, int W, int H1, int W1) {
  int idx = blockIdx.x * blockDim.x + threadIdx.x;
  int HW = H * W;
  if (idx >= D * HW) return;
  int d = idx / HW;
  int pix = idx - d * HW;
  int h = pix / W;
  int w = pix - h * W;

  float s0, sstep;
  samp_base_step<STAGE>(h, w, dv, d1, v1, H1, W1, s0, sstep);
  float samp = s0 + sstep * (float)d;

  int base[2][4];
  float wt[2][4];
#pragma unroll
  for (int v = 0; v < 2; ++v) {
    const float* R = rt + v * 12;
    float x = (float)w, y = (float)h;
    float rx = R[0] * x + R[1] * y + R[2];
    float ry = R[3] * x + R[4] * y + R[5];
    float rz = R[6] * x + R[7] * y + R[8];
    float px = rx * samp + R[9];
    float py = ry * samp + R[10];
    float pz = rz * samp + R[11];
    float gx = (px / pz) / ((W - 1) * 0.5f) - 1.f;
    float gy = (py / pz) / ((H - 1) * 0.5f) - 1.f;
    float ix = ((gx + 1.f) * W - 1.f) * 0.5f;
    float iy = ((gy + 1.f) * H - 1.f) * 0.5f;
    float x0f = floorf(ix), y0f = floorf(iy);
    float fx = ix - x0f, fy = iy - y0f;
    int x0 = (int)x0f, y0 = (int)y0f;
#pragma unroll
    for (int t = 0; t < 4; ++t) {
      int xi = x0 + (t & 1);
      int yi = y0 + (t >> 1);
      bool valid = (xi >= 0) && (xi <= W - 1) && (yi >= 0) && (yi <= H - 1);
      int xc = min(max(xi, 0), W - 1);
      int yc = min(max(yi, 0), H - 1);
      base[v][t] = yc * W + xc;
      float ww = ((t & 1) ? fx : (1.f - fx)) * ((t >> 1) ? fy : (1.f - fy));
      wt[v][t] = valid ? ww : 0.f;
    }
  }

  const float* f0 = feats;               // ref view
  const float* f1 = feats + (size_t)1 * C * HW;
  const float* f2 = feats + (size_t)2 * C * HW;
  const float inv3 = 1.f / 3.f;

  union { _Float16 h[C]; u32x4 q[C / 8]; } row;
#pragma unroll 8
  for (int c = 0; c < C; ++c) {
    size_t co = (size_t)c * HW;
    float r = f0[co + pix];
    float vsum = r, vsq = r * r;
    float w1v = wt[0][0] * f1[co + base[0][0]] + wt[0][1] * f1[co + base[0][1]] +
                wt[0][2] * f1[co + base[0][2]] + wt[0][3] * f1[co + base[0][3]];
    vsum += w1v; vsq += w1v * w1v;
    float w2v = wt[1][0] * f2[co + base[1][0]] + wt[1][1] * f2[co + base[1][1]] +
                wt[1][2] * f2[co + base[1][2]] + wt[1][3] * f2[co + base[1][3]];
    vsum += w2v; vsq += w2v * w2v;
    float m = vsum * inv3;
    row.h[c] = (_Float16)(vsq * inv3 - m * m);
  }

  u32x4* vb = reinterpret_cast<u32x4*>(varBuf + (size_t)idx * C);
#pragma unroll
  for (int i = 0; i < C / 8; ++i) vb[i] = row.q[i];
}

// ----------------- K2: conv3d (C->1, 3x3x3, SAME) via WMMA -----------------
// One wave per 16-voxel W-tile. 27 taps, each one v_wmma_f32_16x16x32_f16.
// A: 16 voxels x C channels (documented 16-bit A layout: lane<16 holds
// K {0..7,16..23} as two contiguous 8-half runs -> two 16B loads).
// B: f16 weight vector from the packed [tap][K] table, broadcast across all
// N columns (layout-robust: every D column equals the conv output).

template <int C>
__global__ void mvs_conv3d_wmma_kernel(const _Float16* __restrict__ varBuf,
                                       const _Float16* __restrict__ wgtH,  // [27][32]
                                       const float* __restrict__ bias,     // (1,)
                                       float* __restrict__ score,
                                       int D, int H, int W) {
  int lane = threadIdx.x & 31;
  int wave = (blockIdx.x * blockDim.x + threadIdx.x) >> 5;
  int tilesPerRow = W >> 4;
  int w0 = (wave % tilesPerRow) << 4;
  int row = wave / tilesPerRow;
  int h = row % H;
  int dz = row / H;

  int myw = w0 + (lane & 15);     // this lane's voxel (A-matrix row M)
  int k0 = (lane < 16) ? 0 : 8;   // A: first contiguous K run base
  int kb = (lane < 16) ? 0 : 8;   // B: first contiguous K run base (same as A)

  v8f acc = (v8f)(0.f);
#pragma unroll
  for (int kd = -1; kd <= 1; ++kd) {
#pragma unroll
    for (int kh = -1; kh <= 1; ++kh) {
#pragma unroll
      for (int kw = -1; kw <= 1; ++kw) {
        int dz2 = dz + kd, hy = h + kh, wx = myw + kw;
        bool inb = (dz2 >= 0) & (dz2 < D) & (hy >= 0) & (hy < H) & (wx >= 0) & (wx < W);

        union { v16h v; u32x4 q[2]; } A;
        A.q[0] = (u32x4)(0u);
        A.q[1] = (u32x4)(0u);
        if (inb) {
          const _Float16* vp = varBuf + ((size_t)(dz2 * H + hy) * W + wx) * C;
          A.q[0] = *reinterpret_cast<const u32x4*>(vp + k0);          // K k0..k0+7
          if (C > 16)
            A.q[1] = *reinterpret_cast<const u32x4*>(vp + k0 + 16);   // K k0+16..k0+23
        }

        int tap = (kd + 1) * 9 + (kh + 1) * 3 + (kw + 1);
        union { v16h v; u32x4 q[2]; } B;
        const _Float16* wp = wgtH + tap * 32 + kb;
        B.q[0] = *reinterpret_cast<const u32x4*>(wp);      // K kb..kb+7
        B.q[1] = *reinterpret_cast<const u32x4*>(wp + 16);  // K kb+16..kb+23

        acc = __builtin_amdgcn_wmma_f32_16x16x32_f16(
            false, A.v, false, B.v, (short)0, acc, false, false);
      }
    }
  }

  union { v8f v; float f[8]; } Dd;
  Dd.v = acc;
  float b = bias[0];
  size_t obase = (size_t)(dz * H + h) * W + w0;
  if (lane == 0) {
#pragma unroll
    for (int r = 0; r < 8; ++r) score[obase + r] = Dd.f[r] + b;       // M=0..7
  } else if (lane == 16) {
#pragma unroll
    for (int r = 0; r < 8; ++r) score[obase + 8 + r] = Dd.f[r] + b;   // M=8..15
  }
}

// ------------------------ K3: in-place softmax over D ----------------------

__global__ void mvs_softmax_kernel(float* __restrict__ buf, int Dn, int HW) {
  int i = blockIdx.x * blockDim.x + threadIdx.x;
  if (i >= HW) return;
  float m = -3.4e38f;
  for (int d = 0; d < Dn; ++d) m = fmaxf(m, buf[(size_t)d * HW + i]);
  float s = 0.f;
  for (int d = 0; d < Dn; ++d) {
    float e = expf(buf[(size_t)d * HW + i] - m);
    buf[(size_t)d * HW + i] = e;
    s += e;
  }
  float inv = 1.f / s;
  for (int d = 0; d < Dn; ++d) buf[(size_t)d * HW + i] *= inv;
}

// ------------- K4: expectation depth, confidence, expected variance --------

template <int STAGE>
__global__ void mvs_stats_kernel(const float* __restrict__ prob,
                                 const float* __restrict__ dv,
                                 const float* __restrict__ d1,
                                 const float* __restrict__ v1, int H1, int W1,
                                 float* __restrict__ dOut, float* __restrict__ cOut,
                                 float* __restrict__ vOut, int Dn, int H, int W) {
  int pix = blockIdx.x * blockDim.x + threadIdx.x;
  int HW = H * W;
  if (pix >= HW) return;
  int h = pix / W, w = pix - (pix / W) * W;

  float s0, sstep;
  samp_base_step<STAGE>(h, w, dv, d1, v1, H1, W1, s0, sstep);

  float depth = 0.f, dif = 0.f;
  for (int d = 0; d < Dn; ++d) {
    float p = prob[(size_t)d * HW + pix];
    depth += p * (s0 + sstep * (float)d);
    dif += p * (float)d;
  }
  int di = min(max((int)dif, 0), Dn - 1);

  float conf = 0.f, var = 0.f;
  for (int d = 0; d < Dn; ++d) {
    float p = prob[(size_t)d * HW + pix];
    float s = s0 + sstep * (float)d;
    var += (s - depth) * (s - depth) * p;
    if (d >= di - 1 && d <= di + 2) conf += p;  // sum4 window at di
  }
  dOut[pix] = depth;
  cOut[pix] = conf;
  vOut[pix] = LAMB * sqrtf(var);
}

// ------------------------------- launcher ---------------------------------

extern "C" void kernel_launch(void* const* d_in, const int* in_sizes, int n_in,
                              void* d_out, int out_size, void* d_ws, size_t ws_size,
                              hipStream_t stream) {
  (void)in_sizes; (void)n_in; (void)out_size; (void)ws_size;
  const float* feats1 = (const float*)d_in[0];  // (1,3,32,128,160)
  const float* feats2 = (const float*)d_in[1];  // (1,3,16,256,320)
  const float* proj1  = (const float*)d_in[2];  // (1,3,2,4,4)
  const float* proj2  = (const float*)d_in[3];  // (1,3,2,4,4)
  const float* dv     = (const float*)d_in[4];  // (1,48)
  // d_in[5] = img (shape only, values unused by the math we need)
  const float* w1 = (const float*)d_in[6];      // (1,32,3,3,3)
  const float* b1 = (const float*)d_in[7];
  const float* w2 = (const float*)d_in[8];      // (1,16,3,3,3)
  const float* b2 = (const float*)d_in[9];

  // workspace: [rt 256B][wgtH 3456B -> up to 4KB][score/prob 4MB][f16 variance ~63MB]
  float* rt = (float*)d_ws;
  _Float16* wgtH = (_Float16*)((char*)d_ws + 256);
  float* score = (float*)((char*)d_ws + 8192);
  _Float16* varBuf = (_Float16*)((char*)d_ws + 8192 + (size_t)4 * 1024 * 1024);

  float* out = (float*)d_out;
  float* d1o = out;                 // (1,128,160)
  float* c1o = out + 20480;
  float* v1o = out + 40960;
  float* d2o = out + 61440;         // (1,256,320)
  float* c2o = out + 143360;
  float* v2o = out + 225280;

  mvs_compose_kernel<<<1, 32, 0, stream>>>(proj1, proj2, rt);
  mvs_pack_weights_kernel<<<(2 * 27 * 32 + 255) / 256, 256, 0, stream>>>(w1, w2, wgtH);

  // ---- stage 1: C=32, D=48, H=128, W=160 ----
  {
    const int D = ND1, H = 128, W = 160, HW = H * W;
    mvs_warp_var_kernel<32, 1><<<(D * HW) / 256, 256, 0, stream>>>(
        feats1, rt, dv, nullptr, nullptr, varBuf, D, H, W, 0, 0);
    int tiles = D * H * (W / 16);  // 61440, one wave each; 4 waves/block
    mvs_conv3d_wmma_kernel<32><<<tiles / 4, 128, 0, stream>>>(
        varBuf, wgtH, b1, score, D, H, W);
    mvs_softmax_kernel<<<(HW + 255) / 256, 256, 0, stream>>>(score, D, HW);
    mvs_stats_kernel<1><<<(HW + 255) / 256, 256, 0, stream>>>(
        score, dv, nullptr, nullptr, 0, 0, d1o, c1o, v1o, D, H, W);
  }

  // ---- stage 2: C=16, D=8, H=256, W=320 (samples from d1,v1 upsampled) ----
  {
    const int D = ND2, H = 256, W = 320, HW = H * W;
    mvs_warp_var_kernel<16, 2><<<(D * HW) / 256, 256, 0, stream>>>(
        feats2, rt + 24, dv, d1o, v1o, varBuf, D, H, W, 128, 160);
    int tiles = D * H * (W / 16);  // 40960
    mvs_conv3d_wmma_kernel<16><<<tiles / 4, 128, 0, stream>>>(
        varBuf, wgtH + 27 * 32, b2, score, D, H, W);
    mvs_softmax_kernel<<<(HW + 255) / 256, 256, 0, stream>>>(score, D, HW);
    mvs_stats_kernel<2><<<(HW + 255) / 256, 256, 0, stream>>>(
        score, dv, d1o, v1o, 128, 160, d2o, c2o, v2o, D, H, W);
  }
}
